// PointNet2_75084618268927
// MI455X (gfx1250) — compile-verified, matrix operand order checked by reading, not run on hardware
//
#include <hip/hip_runtime.h>

// PointNet++ part-segmentation forward for MI455X (gfx1250, wave32).
// GEMMs: v_wmma_f32_16x16x32_f16, f16 activations/weights staged so each
// lane's fragment is two contiguous 16B runs -> global_load_b128, no guards.

#define SLOPE 0.2f

typedef __attribute__((ext_vector_type(16))) _Float16 v16h;
typedef __attribute__((ext_vector_type(8)))  _Float16 h8;
typedef __attribute__((ext_vector_type(8)))  float    v8f;

__device__ __forceinline__ void atomicMaxFloat(float* addr, float v) {
  if (v >= 0.f) atomicMax((int*)addr, __float_as_int(v));
  else          atomicMin((unsigned int*)addr, __float_as_uint(v));
}

__global__ void k_fill(float* p, int n, float v) {
  int t = blockIdx.x * blockDim.x + threadIdx.x;
  if (t < n) p[t] = v;
}

__global__ void k_h2f(const _Float16* __restrict__ in, float* __restrict__ out, int n) {
  int t = blockIdx.x * blockDim.x + threadIdx.x;
  if (t < n) out[t] = (float)in[t];
}

// Pack W (cin x cout, f32) -> Wt (cout x kpad, f16), zero-padded K tail.
__global__ void k_packw(const float* __restrict__ W, int cin, int cout, int kpad,
                        _Float16* __restrict__ Wt, int total) {
  int t = blockIdx.x * blockDim.x + threadIdx.x;
  if (t >= total) return;
  int c = t / kpad, k = t % kpad;
  Wt[t] = (k < cin) ? (_Float16)W[(size_t)k * cout + c] : (_Float16)0.f;
}

// -------- Farthest point sampling: one block per batch, dist in LDS --------
__global__ __launch_bounds__(256) void k_fps(const float* __restrict__ xyz,
                                             int n, int stride, int s,
                                             int* __restrict__ idx_out) {
  int b = blockIdx.x;
  const float* p = xyz + (size_t)b * n * stride;
  __shared__ float dist[4096];
  __shared__ float rmax[256];
  __shared__ int   rarg[256];
  __shared__ int   s_far;
  int tid = threadIdx.x;
  for (int i = tid; i < n; i += 256) dist[i] = 1e10f;
  if (tid == 0) s_far = 0;
  __syncthreads();
  for (int it = 0; it < s; ++it) {
    int far = s_far;
    if (tid == 0) idx_out[b * s + it] = far;
    float cx = p[far * stride + 0], cy = p[far * stride + 1], cz = p[far * stride + 2];
    float lmax = -1.f; int larg = 0;
    for (int i = tid; i < n; i += 256) {
      float dx = p[i * stride + 0] - cx;
      float dy = p[i * stride + 1] - cy;
      float dz = p[i * stride + 2] - cz;
      float d = dx * dx + dy * dy + dz * dz;
      float dd = fminf(dist[i], d);
      dist[i] = dd;
      if (dd > lmax) { lmax = dd; larg = i; }
    }
    rmax[tid] = lmax; rarg[tid] = larg;
    __syncthreads();
    for (int off = 128; off > 0; off >>= 1) {
      if (tid < off && rmax[tid + off] > rmax[tid]) {
        rmax[tid] = rmax[tid + off]; rarg[tid] = rarg[tid + off];
      }
      __syncthreads();
    }
    if (tid == 0) s_far = rarg[0];
    __syncthreads();
  }
}

__global__ void k_gather3(const float* __restrict__ xyz, int n, int stride,
                          const int* __restrict__ idx, int s,
                          float* __restrict__ out, int total) {
  int t = blockIdx.x * blockDim.x + threadIdx.x;
  if (t >= total) return;
  int b = t / s;
  int pi = idx[t];
  const float* src = xyz + ((size_t)b * n + pi) * stride;
  out[t * 3 + 0] = src[0];
  out[t * 3 + 1] = src[1];
  out[t * 3 + 2] = src[2];
}

// -------- Ball query: first k indices within radius, pad with first --------
__global__ void k_ball(const float* __restrict__ xyz, int n, int stride,
                       const float* __restrict__ centers, int s,
                       float r2, int k, int* __restrict__ idx_out, int total) {
  int t = blockIdx.x * blockDim.x + threadIdx.x;
  if (t >= total) return;
  int b = t / s;
  const float* p = xyz + (size_t)b * n * stride;
  float cx = centers[t * 3 + 0], cy = centers[t * 3 + 1], cz = centers[t * 3 + 2];
  int* out = idx_out + (size_t)t * k;
  int cnt = 0, first = 0;
  for (int i = 0; i < n && cnt < k; ++i) {
    float dx = p[i * stride + 0] - cx;
    float dy = p[i * stride + 1] - cy;
    float dz = p[i * stride + 2] - cz;
    float d2 = dx * dx + dy * dy + dz * dz;
    if (d2 < r2) { if (cnt == 0) first = i; out[cnt++] = i; }
  }
  int fill = cnt ? first : 0;
  for (int j = cnt; j < k; ++j) out[j] = fill;
}

// -------- Group -> f16 rows, K-padded: [xyz[pi]-center, feats[pi], 0...] ----
__global__ void k_group_h(const float* __restrict__ xyz, int n, int stx,
                          const float* __restrict__ feats, int cf, int stf,
                          const float* __restrict__ centers,
                          const int* __restrict__ idx, int s, int k,
                          _Float16* __restrict__ out, int ldo, int total) {
  int t = blockIdx.x * blockDim.x + threadIdx.x;
  if (t >= total) return;
  int cs = t / k;
  int b = cs / s;
  int pi = idx[t];
  const float* pp = xyz + ((size_t)b * n + pi) * stx;
  const float* cc = centers + (size_t)cs * 3;
  _Float16* o = out + (size_t)t * ldo;
  o[0] = (_Float16)(pp[0] - cc[0]);
  o[1] = (_Float16)(pp[1] - cc[1]);
  o[2] = (_Float16)(pp[2] - cc[2]);
  const float* ff = feats + ((size_t)b * n + pi) * stf;
  for (int c = 0; c < cf; ++c) o[3 + c] = (_Float16)ff[c];
  for (int c = 3 + cf; c < ldo; ++c) o[c] = (_Float16)0.f;
}

// -------- Row-wise concat -> f16 padded --------
__global__ void k_concat_h(const float* __restrict__ A, int ca,
                           const float* __restrict__ Bf, int cb,
                           _Float16* __restrict__ out, int ldo, int rows) {
  int t = blockIdx.x * blockDim.x + threadIdx.x;
  if (t >= rows) return;
  _Float16* o = out + (size_t)t * ldo;
  const float* a = A + (size_t)t * ca;
  const float* b = Bf + (size_t)t * cb;
  for (int c = 0; c < ca; ++c) o[c] = (_Float16)a[c];
  for (int c = 0; c < cb; ++c) o[ca + c] = (_Float16)b[c];
  for (int c = ca + cb; c < ldo; ++c) o[c] = (_Float16)0.f;
}

// -------- 3-NN inverse-distance interp + concat -> f16 padded --------
__global__ void k_interp_h(const float* __restrict__ xyz1, int n1, int st1,
                           const float* __restrict__ xyz2, int n2,
                           const float* __restrict__ f2, int c2,
                           const float* __restrict__ f1, int c1, int stf1,
                           _Float16* __restrict__ out, int ldo, int total) {
  int t = blockIdx.x * blockDim.x + threadIdx.x;
  if (t >= total) return;
  int b = t / n1, i = t % n1;
  const float* p = xyz1 + ((size_t)b * n1 + i) * st1;
  float px = p[0], py = p[1], pz = p[2];
  const float* q = xyz2 + (size_t)b * n2 * 3;
  float bd0 = 1e30f, bd1 = 1e30f, bd2 = 1e30f;
  int bi0 = 0, bi1 = 0, bi2 = 0;
  for (int j = 0; j < n2; ++j) {
    float dx = q[j * 3 + 0] - px;
    float dy = q[j * 3 + 1] - py;
    float dz = q[j * 3 + 2] - pz;
    float d = dx * dx + dy * dy + dz * dz;
    if (d < bd0)      { bd2 = bd1; bi2 = bi1; bd1 = bd0; bi1 = bi0; bd0 = d; bi0 = j; }
    else if (d < bd1) { bd2 = bd1; bi2 = bi1; bd1 = d; bi1 = j; }
    else if (d < bd2) { bd2 = d; bi2 = j; }
  }
  if (n2 < 2) { bd1 = bd0; bi1 = bi0; }
  if (n2 < 3) { bd2 = bd0; bi2 = bi0; }
  float w0 = 1.f / fmaxf(bd0, 1e-10f);
  float w1 = 1.f / fmaxf(bd1, 1e-10f);
  float w2 = 1.f / fmaxf(bd2, 1e-10f);
  float ws = w0 + w1 + w2;
  w0 /= ws; w1 /= ws; w2 /= ws;
  const float* fb = f2 + (size_t)b * n2 * c2;
  _Float16* o = out + (size_t)t * ldo;
  for (int c = 0; c < c2; ++c)
    o[c] = (_Float16)(w0 * fb[(size_t)bi0 * c2 + c] + w1 * fb[(size_t)bi1 * c2 + c] +
                      w2 * fb[(size_t)bi2 * c2 + c]);
  const float* fp = f1 + ((size_t)b * n1 + i) * stf1;
  for (int c = 0; c < c1; ++c) o[c2 + c] = (_Float16)fp[c];
  for (int c = c2 + c1; c < ldo; ++c) o[c] = (_Float16)0.f;
}

// -------- WMMA MLP layer: Y = act(X @ W + b) ------------------------------
// X: f16, row stride ldx (mult of 32). Wt: f16, transposed, cout x kpad.
// Wave computes 32x16 output (two M-tiles sharing the B fragment):
// per K-step 6x global_load_b128 + 2x v_wmma_f32_16x16x32_f16, no guards.
// Epilogue: f16 store / f32 store / fused atomic max-pool over pool_k rows.
__global__ __launch_bounds__(256)
void k_mlp_wmma(const _Float16* __restrict__ X, int ldx,
                const _Float16* __restrict__ Wt, int kpad,
                const float* __restrict__ bias, void* __restrict__ Yv,
                int M, int Cout, int act, int pool_k, int out_half) {
  int Mt2 = M >> 5;                      // M is always a multiple of 32 here
  int Nt = (Cout + 15) >> 4;
  int tiles = Mt2 * Nt;
  int wave = threadIdx.x >> 5;
  int lane = threadIdx.x & 31;
  int t = blockIdx.x * 8 + wave;
  if (t >= tiles) return;                // wave-uniform exit
  int tm = t % Mt2, tn = t / Mt2;
  int hi = lane >> 4;
  int lm = lane & 15;
  int row0 = tm * 32 + lm;
  int coln = tn * 16 + lm;
  int colc = coln < Cout ? coln : Cout - 1;   // clamp loads; store is guarded
  const _Float16* xp0 = X + (size_t)row0 * ldx + (hi ? 8 : 0);
  const _Float16* xp1 = xp0 + (size_t)16 * ldx;
  const _Float16* wp  = Wt + (size_t)colc * kpad + (hi ? 8 : 0);
  v8f acc0 = {}, acc1 = {};
  for (int k0 = 0; k0 < kpad; k0 += 32) {
    h8 a0l = *(const h8*)(xp0 + k0);
    h8 a0h = *(const h8*)(xp0 + k0 + 16);
    h8 a1l = *(const h8*)(xp1 + k0);
    h8 a1h = *(const h8*)(xp1 + k0 + 16);
    h8 bl  = *(const h8*)(wp + k0);
    h8 bh  = *(const h8*)(wp + k0 + 16);
    v16h a0 = __builtin_shufflevector(a0l, a0h, 0,1,2,3,4,5,6,7,8,9,10,11,12,13,14,15);
    v16h a1 = __builtin_shufflevector(a1l, a1h, 0,1,2,3,4,5,6,7,8,9,10,11,12,13,14,15);
    v16h bb = __builtin_shufflevector(bl,  bh,  0,1,2,3,4,5,6,7,8,9,10,11,12,13,14,15);
    acc0 = __builtin_amdgcn_wmma_f32_16x16x32_f16(false, a0, false, bb,
                                                  (short)0, acc0, false, false);
    acc1 = __builtin_amdgcn_wmma_f32_16x16x32_f16(false, a1, false, bb,
                                                  (short)0, acc1, false, false);
  }
  int nn = tn * 16 + lm;
  float bv = bias[colc];
#pragma unroll
  for (int r = 0; r < 8; ++r) {
    int m0 = tm * 32 + r + (hi ? 8 : 0);
    int m1 = m0 + 16;
    float v0 = acc0[r] + bv;
    float v1 = acc1[r] + bv;
    if (act) {
      v0 = v0 > 0.f ? v0 : SLOPE * v0;
      v1 = v1 > 0.f ? v1 : SLOPE * v1;
    }
    if (nn < Cout) {
      if (out_half) {
        ((_Float16*)Yv)[(size_t)m0 * Cout + nn] = (_Float16)v0;
        ((_Float16*)Yv)[(size_t)m1 * Cout + nn] = (_Float16)v1;
      } else if (pool_k > 0) {
        atomicMaxFloat(&((float*)Yv)[(size_t)(m0 / pool_k) * Cout + nn], v0);
        atomicMaxFloat(&((float*)Yv)[(size_t)(m1 / pool_k) * Cout + nn], v1);
      } else {
        ((float*)Yv)[(size_t)m0 * Cout + nn] = v0;
        ((float*)Yv)[(size_t)m1 * Cout + nn] = v1;
      }
    }
  }
}

// ===========================================================================
static const int LCIN[25]  = {9,32,32, 67,64,64, 131,128,128, 259,256,256,
                              768,256, 384,256, 320,256, 137,128,128,128,
                              128,128,64};
static const int LCOUT[25] = {32,32,64, 64,64,128, 128,128,256, 256,256,512,
                              256,256, 256,256, 256,128, 128,128,128,128,
                              128,64,13};

extern "C" void kernel_launch(void* const* d_in, const int* in_sizes, int n_in,
                              void* d_out, int out_size, void* d_ws, size_t ws_size,
                              hipStream_t stream) {
  (void)in_sizes; (void)n_in; (void)out_size; (void)ws_size;
  const int Bb = 8, Nn = 4096;
  const float* pc = (const float*)d_in[0];          // (8,4096,9)
  float* outp = (float*)d_out;
  const size_t PF_OFF = (size_t)Bb * Nn * 128;
  const size_t GF_OFF = PF_OFF + (size_t)Bb * 512;

  // ---- workspace carve ----
  char* wsp = (char*)d_ws;
  size_t off = 0;
  auto alloc = [&](size_t bytes) -> void* {
    void* p = wsp + off;
    off += (bytes + 255) & ~(size_t)255;
    return p;
  };
  float* f1   = (float*)alloc((size_t)Bb * 1024 * 64  * 4);
  float* f2   = (float*)alloc((size_t)Bb * 256  * 128 * 4);
  float* f3   = (float*)alloc((size_t)Bb * 64   * 256 * 4);
  float* f4   = (float*)alloc((size_t)Bb * 512        * 4);
  float* f3n  = (float*)alloc((size_t)Bb * 64   * 256 * 4);
  float* f2n  = (float*)alloc((size_t)Bb * 256  * 256 * 4);
  float* f1n  = (float*)alloc((size_t)Bb * 1024 * 128 * 4);
  float* xyz1 = (float*)alloc((size_t)Bb * 1024 * 3 * 4);
  float* xyz2 = (float*)alloc((size_t)Bb * 256  * 3 * 4);
  float* xyz3 = (float*)alloc((size_t)Bb * 64   * 3 * 4);
  int*   ifps = (int*)  alloc((size_t)Bb * 1024 * 4);
  int*   ibal = (int*)  alloc((size_t)Bb * 1024 * 32 * 4);
  _Float16* wts = (_Float16*)alloc((size_t)1000448 * 2);     // packed weights
  _Float16* arA = (_Float16*)alloc((size_t)12582912 * 2);    // SA2 grouped 131072x96
  _Float16* arB = (_Float16*)alloc((size_t)8388608 * 2);

  auto cdiv = [](int a, int b) { return (a + b - 1) / b; };
  auto rnd32 = [](int x) { return (x + 31) & ~31; };
  auto fill = [&](float* p, int n, float v) {
    k_fill<<<cdiv(n, 256), 256, 0, stream>>>(p, n, v);
  };

  // ---- pack all 25 weight matrices to transposed/padded f16 ----
  size_t woff[25];
  {
    size_t o = 0;
    for (int li = 0; li < 25; ++li) {
      woff[li] = o;
      int kpad = rnd32(LCIN[li]);
      int n = kpad * LCOUT[li];
      k_packw<<<cdiv(n, 256), 256, 0, stream>>>(
          (const float*)d_in[1 + 2 * li], LCIN[li], LCOUT[li], kpad, wts + o, n);
      o += (size_t)n;
    }
  }

  auto mlp = [&](const _Float16* Xh, int ldx, int li, int M, void* Y,
                 int act, int pool, int out_half) {
    int cout = LCOUT[li];
    int kpad = rnd32(LCIN[li]);
    int tiles = (M >> 5) * cdiv(cout, 16);
    k_mlp_wmma<<<cdiv(tiles, 8), 256, 0, stream>>>(
        Xh, ldx, wts + woff[li], kpad, (const float*)d_in[2 + 2 * li],
        Y, M, cout, act, pool, out_half);
  };

  // ================= SA1: 4096 -> 1024, r=0.1, k=32, MLP 9->32->32->64 ====
  k_fps<<<Bb, 256, 0, stream>>>(pc, Nn, 9, 1024, ifps);
  k_gather3<<<cdiv(Bb * 1024, 256), 256, 0, stream>>>(pc, Nn, 9, ifps, 1024, xyz1, Bb * 1024);
  k_ball<<<cdiv(Bb * 1024, 256), 256, 0, stream>>>(pc, Nn, 9, xyz1, 1024, 0.01f, 32, ibal, Bb * 1024);
  {
    int tot = Bb * 1024 * 32;
    k_group_h<<<cdiv(tot, 256), 256, 0, stream>>>(pc, Nn, 9, pc + 3, 6, 9, xyz1, ibal, 1024, 32, arA, 32, tot);
    mlp(arA, 32, 0, tot, arB, 1, 0, 1);
    mlp(arB, 32, 1, tot, arA, 1, 0, 1);
    fill(f1, Bb * 1024 * 64, -1e30f);
    mlp(arA, 32, 2, tot, f1, 1, 32, 0);        // fused max over k=32
  }

  // ================= SA2: 1024 -> 256, r=0.2, k=64, MLP 67->64->64->128 ===
  k_fps<<<Bb, 256, 0, stream>>>(xyz1, 1024, 3, 256, ifps);
  k_gather3<<<cdiv(Bb * 256, 256), 256, 0, stream>>>(xyz1, 1024, 3, ifps, 256, xyz2, Bb * 256);
  k_ball<<<cdiv(Bb * 256, 256), 256, 0, stream>>>(xyz1, 1024, 3, xyz2, 256, 0.04f, 64, ibal, Bb * 256);
  {
    int tot = Bb * 256 * 64;
    k_group_h<<<cdiv(tot, 256), 256, 0, stream>>>(xyz1, 1024, 3, f1, 64, 64, xyz2, ibal, 256, 64, arA, 96, tot);
    mlp(arA, 96, 3, tot, arB, 1, 0, 1);
    mlp(arB, 64, 4, tot, arA, 1, 0, 1);
    fill(f2, Bb * 256 * 128, -1e30f);
    mlp(arA, 64, 5, tot, f2, 1, 64, 0);
  }

  // ================= SA3: 256 -> 64, r=0.4, k=128, MLP 131->128->128->256 =
  k_fps<<<Bb, 256, 0, stream>>>(xyz2, 256, 3, 64, ifps);
  k_gather3<<<cdiv(Bb * 64, 256), 256, 0, stream>>>(xyz2, 256, 3, ifps, 64, xyz3, Bb * 64);
  k_ball<<<cdiv(Bb * 64, 256), 256, 0, stream>>>(xyz2, 256, 3, xyz3, 64, 0.16f, 128, ibal, Bb * 64);
  {
    int tot = Bb * 64 * 128;
    k_group_h<<<cdiv(tot, 256), 256, 0, stream>>>(xyz2, 256, 3, f2, 128, 128, xyz3, ibal, 64, 128, arA, 160, tot);
    mlp(arA, 160, 6, tot, arB, 1, 0, 1);
    mlp(arB, 128, 7, tot, arA, 1, 0, 1);
    fill(f3, Bb * 64 * 256, -1e30f);
    mlp(arA, 128, 8, tot, f3, 1, 128, 0);
  }

  // ================= SA_all: MLP 259->256->256->512, max over 64 ==========
  {
    int rows = Bb * 64;
    k_concat_h<<<cdiv(rows, 256), 256, 0, stream>>>(xyz3, 3, f3, 256, arA, 288, rows);
    mlp(arA, 288, 9, rows, arB, 1, 0, 1);
    mlp(arB, 256, 10, rows, arA, 1, 0, 1);
    fill(f4, Bb * 512, -1e30f);
    mlp(arA, 256, 11, rows, f4, 1, 64, 0);     // pool over all 64 points
  }

  // ================= FP3: 64 <- 1 (broadcast f4), MLP 768->256->256 =======
  {
    int tot = Bb * 64;
    k_interp_h<<<cdiv(tot, 256), 256, 0, stream>>>(xyz3, 64, 3, xyz3, 1, f4, 512, f3, 256, 256, arA, 768, tot);
    mlp(arA, 768, 12, tot, arB, 1, 0, 1);
    mlp(arB, 256, 13, tot, f3n, 1, 0, 0);
  }
  // ================= FP2: 256 <- 64, MLP 384->256->256 ====================
  {
    int tot = Bb * 256;
    k_interp_h<<<cdiv(tot, 256), 256, 0, stream>>>(xyz2, 256, 3, xyz3, 64, f3n, 256, f2, 128, 128, arA, 384, tot);
    mlp(arA, 384, 14, tot, arB, 1, 0, 1);
    mlp(arB, 256, 15, tot, f2n, 1, 0, 0);
  }
  // ================= FP1: 1024 <- 256, MLP 320->256->128 ==================
  {
    int tot = Bb * 1024;
    k_interp_h<<<cdiv(tot, 256), 256, 0, stream>>>(xyz1, 1024, 3, xyz2, 256, f2n, 256, f1, 64, 64, arA, 320, tot);
    mlp(arA, 320, 16, tot, arB, 1, 0, 1);
    mlp(arB, 256, 17, tot, f1n, 1, 0, 0);
  }
  // ======== FP0: 4096 <- 1024, MLP 137->128->128->128->128 ================
  {
    int tot = Bb * Nn;
    k_interp_h<<<cdiv(tot, 256), 256, 0, stream>>>(pc, Nn, 9, xyz1, 1024, f1n, 128, pc, 9, 9, arA, 160, tot);
    mlp(arA, 160, 18, tot, arB, 1, 0, 1);
    mlp(arB, 128, 19, tot, arA, 1, 0, 1);
    mlp(arA, 128, 20, tot, arB, 1, 0, 1);
    mlp(arB, 128, 21, tot, arA, 1, 0, 1);      // point features (f16)
    k_h2f<<<cdiv(tot * 128, 256), 256, 0, stream>>>(arA, outp, tot * 128);
  }

  // ================= global features =================
  hipMemcpyAsync(outp + PF_OFF, f4, (size_t)Bb * 512 * 4,
                 hipMemcpyDeviceToDevice, stream);

  // ================= segmentation head: 128->128->64->13 =================
  {
    int tot = Bb * Nn;
    mlp(arA, 128, 22, tot, arB, 1, 0, 1);
    mlp(arB, 128, 23, tot, arA, 1, 0, 1);
    mlp(arA, 64, 24, tot, outp + GF_OFF, 0, 0, 0);   // logits, no activation
  }
}
